// LabelSmoothing_42176578846915
// MI455X (gfx1250) — matrix operand order repaired and checked by reading, not compiled, not run on hardware
//
#include <hip/hip_runtime.h>
#include <hip/hip_bf16.h>
#include <math.h>

typedef __attribute__((ext_vector_type(16))) _Float16 v16h;
typedef __attribute__((ext_vector_type(8)))  float    v8f;

#define TILE_R 16
#define TILE_C 128
#define HALO   5
#define ER     (TILE_R + 2*HALO)   // 26 rows staged
#define EC     (TILE_C + 2*HALO)   // 138 cols staged
#define IMG_H  1024
#define IMG_W  1024
#define NBATCH 8
#define NPIX   (NBATCH * IMG_H * IMG_W)

// Fused: stage target tile+halo -> horizontal 11-tap sliding sums (LDS, f16) ->
// vertical 11-tap via v_wmma_f32_16x16x32_f16 (banded-ones A, exact integer math)
// -> edge flags -> coalesced log-softmax NLL over x -> block partial sum.
__global__ __launch_bounds__(256)
void ls_tile_kernel(const float* __restrict__ x,
                    const int*   __restrict__ target,
                    float*       __restrict__ partials)
{
    __shared__ int           s_t[ER][EC];          // target tile + halo
    __shared__ _Float16      s_h[32][TILE_C];      // horizontal sums (rows 26..31 zero)
    __shared__ unsigned char s_edge[TILE_R][TILE_C];
    __shared__ float         s_red[256];

    const int tid = threadIdx.x;
    const int b   = blockIdx.z;
    const int r0  = blockIdx.y * TILE_R;
    const int c0  = blockIdx.x * TILE_C;

    // ---- Phase 1: stage target (zero-padded halo) ----
    const int* tgt_b = target + (size_t)b * IMG_H * IMG_W;
    for (int i = tid; i < ER * EC; i += 256) {
        int rr = i / EC, cc = i % EC;
        int gr = r0 - HALO + rr;
        int gc = c0 - HALO + cc;
        int v = 0;
        if (gr >= 0 && gr < IMG_H && gc >= 0 && gc < IMG_W)
            v = tgt_b[(size_t)gr * IMG_W + gc];
        s_t[rr][cc] = v;
    }
    __syncthreads();

    // ---- Phase 2: horizontal 11-tap sliding sums (exact small ints in f16) ----
    {
        int r   = tid >> 3;          // 0..31 (26 active rows, rest zero-fill)
        int cch = (tid & 7) * 16;    // 16-col chunk
        if (r < ER) {
            int s = 0;
            #pragma unroll
            for (int d = 0; d < 11; ++d) s += s_t[r][cch + d];
            s_h[r][cch] = (_Float16)(float)s;
            #pragma unroll
            for (int c = 1; c < 16; ++c) {
                s += s_t[r][cch + c + 10] - s_t[r][cch + c - 1];
                s_h[r][cch + c] = (_Float16)(float)s;
            }
        } else {
            #pragma unroll
            for (int c = 0; c < 16; ++c) s_h[r][cch + c] = (_Float16)0.0f;
        }
    }
    __syncthreads();

    // ---- Phase 3: vertical 11-tap via WMMA, one 16x16 sub-tile per wave ----
    {
        const int lane = tid & 31;
        const int wcol = (tid >> 5) * 16;   // wave's 16-column slice
        const int nm   = lane & 15;         // M (for A) / N (for B)
        v16h amat, bmat;
        #pragma unroll
        for (int e = 0; e < 16; ++e) {
            // A 16x32 f16 layout: lanes0-15 e<8->K=e, e>=8->K=e+8; lanes16-31 +8
            int ka = ((e < 8) ? e : e + 8) + ((lane < 16) ? 0 : 8);
            amat[e] = (_Float16)((ka >= nm && ka <= nm + 10) ? 1.0f : 0.0f);
            // B 32x16 f16 layout: lanes0-15 e<8->K=e, e>=8->K=e+8; lanes16-31 +8
            int kb = ((e < 8) ? e : e + 8) + ((lane < 16) ? 0 : 8);
            bmat[e] = s_h[kb][wcol + nm];
        }
        v8f dmat = {};
        dmat = __builtin_amdgcn_wmma_f32_16x16x32_f16(
                   false, amat, false, bmat, (short)0, dmat, false, false);
        #pragma unroll
        for (int rI = 0; rI < 8; ++rI) {
            int row = rI + ((lane < 16) ? 0 : 8);   // D: lane half -> M offset
            int col = wcol + nm;
            float box = dmat[rI];                   // exact 11x11 window sum
            int   t   = s_t[row + HALO][col + HALO];
            s_edge[row][col] = (box != (float)(121 * t)) ? 1 : 0;
        }
    }
    __syncthreads();

    // ---- Phase 4: log-softmax NLL, fully coalesced reads of x ----
    const float* x_b0 = x + (size_t)b * 2 * IMG_H * IMG_W;
    const float* x_b1 = x_b0 + (size_t)IMG_H * IMG_W;
    float acc = 0.0f;
    #pragma unroll
    for (int it = 0; it < (TILE_R * TILE_C) / 256; ++it) {
        int p   = tid + it * 256;
        int row = p >> 7;       // /TILE_C
        int col = p & (TILE_C - 1);
        size_t off = (size_t)(r0 + row) * IMG_W + (c0 + col);
        float x0 = x_b0[off];
        float x1 = x_b1[off];
        float m  = fmaxf(x0, x1);
        float l  = m + log1pf(__expf(-fabsf(x0 - x1)));   // logsumexp
        float lp0 = x0 - l;
        float lp1 = x1 - l;
        int   t   = s_t[row + HALO][col + HALO];
        float lpt = t ? lp1 : lp0;
        float lpo = t ? lp0 : lp1;
        acc += s_edge[row][col] ? (0.95f * lpt + 0.1f * lpo) : lpt;
    }

    // ---- Phase 5: block reduction -> one partial ----
    s_red[tid] = acc;
    __syncthreads();
    #pragma unroll
    for (int s = 128; s > 0; s >>= 1) {
        if (tid < s) s_red[tid] += s_red[tid + s];
        __syncthreads();
    }
    if (tid == 0)
        partials[((size_t)blockIdx.z * gridDim.y + blockIdx.y) * gridDim.x + blockIdx.x]
            = s_red[0];
}

__global__ __launch_bounds__(256)
void ls_reduce_kernel(const float* __restrict__ partials,
                      float* __restrict__ out, int n)
{
    __shared__ float s[256];
    float a = 0.0f;
    for (int i = threadIdx.x; i < n; i += 256) a += partials[i];
    s[threadIdx.x] = a;
    __syncthreads();
    #pragma unroll
    for (int st = 128; st > 0; st >>= 1) {
        if (threadIdx.x < st) s[threadIdx.x] += s[threadIdx.x + st];
        __syncthreads();
    }
    if (threadIdx.x == 0) out[0] = -s[0] / (float)NPIX;
}

extern "C" void kernel_launch(void* const* d_in, const int* in_sizes, int n_in,
                              void* d_out, int out_size, void* d_ws, size_t ws_size,
                              hipStream_t stream)
{
    const float* x      = (const float*)d_in[0];   // [8,2,1024,1024] f32
    const int*   target = (const int*)d_in[1];     // [8,1,1024,1024] i32
    float*       out    = (float*)d_out;           // scalar f32
    float*       parts  = (float*)d_ws;            // 4096 partials

    dim3 grid(IMG_W / TILE_C, IMG_H / TILE_R, NBATCH);   // (8, 64, 8) = 4096 blocks
    ls_tile_kernel<<<grid, 256, 0, stream>>>(x, target, parts);

    int nparts = (IMG_W / TILE_C) * (IMG_H / TILE_R) * NBATCH;
    ls_reduce_kernel<<<1, 256, 0, stream>>>(parts, out, nparts);
}
